// SparseMoEMlp_71803263254891
// MI455X (gfx1250) — matrix operand order repaired and checked by reading, not compiled, not run on hardware
//
#include <hip/hip_runtime.h>
#include <math.h>

// ---------------------------------------------------------------------------
// SparseMoE MLP forward for gfx1250 (MI455X, wave32)
//   - WMMA f16 16x16x32, f32 accumulate
//   - B tiles staged global->LDS with async loads (ASYNCcnt), double buffered
// ---------------------------------------------------------------------------

typedef _Float16 h16;
typedef __attribute__((ext_vector_type(16))) _Float16 v16h;
typedef __attribute__((ext_vector_type(8)))  _Float16 v8h;
typedef __attribute__((ext_vector_type(8)))  float    v8f;

#define N_TOK   8192
#define DIM     768
#define HID     3072
#define NE      8
#define TOPK    2
#define CAP     2560                 // round(2*8192*1.25/8)
#define NASSIGN (N_TOK * TOPK)       // 16384
#define NROWS   (NE * CAP)           // 20480

// LDS B-tile geometry: 128 N-rows x 32 K-halves (64B) padded to 80B row stride
#define BROW_B   80                  // bytes per row (64 data + 16 pad)
#define BROW_H   40                  // halves per row
#define BBUF_B   (128 * BROW_B)      // 10240 bytes per buffer
#define BBUF_H   (128 * BROW_H)      // halves per buffer

// ---------------------------------------------------------------------------
// zero fill
// ---------------------------------------------------------------------------
__global__ void k_zero(float* __restrict__ p, int n) {
    int i = blockIdx.x * blockDim.x + threadIdx.x;
    int stride = gridDim.x * blockDim.x;
    for (; i < n; i += stride) p[i] = 0.0f;
}

// ---------------------------------------------------------------------------
// transpose + f32 -> f16 convert:  in [E][R][C]  ->  out [E][C][R]
// ---------------------------------------------------------------------------
__global__ void k_transpose_cvt(const float* __restrict__ in, h16* __restrict__ out,
                                int R, int C) {
    __shared__ h16 tile[32][33];
    int e  = blockIdx.z;
    const float* ine  = in  + (size_t)e * R * C;
    h16*         oute = out + (size_t)e * R * C;
    int c0 = blockIdx.x * 32, r0 = blockIdx.y * 32;
    int tx = threadIdx.x, ty = threadIdx.y;
    #pragma unroll
    for (int i = ty; i < 32; i += 8)
        tile[i][tx] = (h16)ine[(size_t)(r0 + i) * C + (c0 + tx)];
    __syncthreads();
    #pragma unroll
    for (int i = ty; i < 32; i += 8)
        oute[(size_t)(c0 + i) * R + (r0 + tx)] = tile[tx][i];
}

// ---------------------------------------------------------------------------
// router: one wave per token.
// stats[0..7]=importance, [8..15]=load, [16..23]=expert_counts(float),
// ((int*)stats)[24]=dropped (filled by rank kernel)
// ---------------------------------------------------------------------------
__global__ void k_router(const float* __restrict__ x, const float* __restrict__ rw,
                         float* __restrict__ stats,
                         int* __restrict__ ae, float* __restrict__ as_,
                         float* __restrict__ aw) {
    int lane = threadIdx.x & 31;
    int t = blockIdx.x * (blockDim.x >> 5) + (threadIdx.x >> 5);
    if (t >= N_TOK) return;
    const float* xr = x + (size_t)t * DIM;

    float acc[NE];
    #pragma unroll
    for (int e = 0; e < NE; ++e) acc[e] = 0.0f;
    for (int d = lane; d < DIM; d += 32) {
        float xv = xr[d];
        const float* w = rw + (size_t)d * NE;
        #pragma unroll
        for (int e = 0; e < NE; ++e) acc[e] += xv * w[e];
    }
    #pragma unroll
    for (int off = 16; off > 0; off >>= 1) {
        #pragma unroll
        for (int e = 0; e < NE; ++e) acc[e] += __shfl_xor(acc[e], off, 32);
    }
    if (lane != 0) return;

    // softmax over 8 logits
    float mx = acc[0];
    #pragma unroll
    for (int e = 1; e < NE; ++e) mx = fmaxf(mx, acc[e]);
    float p[NE], s = 0.0f;
    #pragma unroll
    for (int e = 0; e < NE; ++e) { p[e] = expf(acc[e] - mx); s += p[e]; }
    float inv = 1.0f / s;
    #pragma unroll
    for (int e = 0; e < NE; ++e) p[e] *= inv;

    // top-2 (lowest index wins ties, like lax.top_k)
    int i1 = 0;
    #pragma unroll
    for (int e = 1; e < NE; ++e) if (acc[e] > acc[i1]) i1 = e;
    int i2 = (i1 == 0) ? 1 : 0;
    #pragma unroll
    for (int e = 0; e < NE; ++e) if (e != i1 && acc[e] > acc[i2]) i2 = e;

    float thr = acc[i2];                 // 2nd-largest logit == sorted[:, E-K]
    float pri = p[i1];                   // clean_probs.max(-1)

    #pragma unroll
    for (int e = 0; e < NE; ++e) {
        atomicAdd(&stats[e], p[e]);                          // importance
        float z = (acc[e] - thr) * 8.0f;                     // / NOISE_STD (=1/8)
        atomicAdd(&stats[8 + e], 0.5f * erfcf(-z * 0.70710678118f)); // P(selected)
    }
    atomicAdd(&stats[16 + i1], 1.0f);
    atomicAdd(&stats[16 + i2], 1.0f);

    int a0 = t * 2;
    ae[a0]     = i1; as_[a0]     = pri;        aw[a0]     = p[i1];
    ae[a0 + 1] = i2; as_[a0 + 1] = pri - 1.0f; aw[a0 + 1] = p[i2];
}

// ---------------------------------------------------------------------------
// exact stable rank-within-expert by counting (LDS-tiled all-pairs)
// ---------------------------------------------------------------------------
__global__ void k_rank(const int* __restrict__ ae, const float* __restrict__ as_,
                       const float* __restrict__ aw,
                       int* __restrict__ dtok, float* __restrict__ wbuf,
                       int* __restrict__ dropped) {
    __shared__ int   se[256];
    __shared__ float ss[256];
    int a = blockIdx.x * blockDim.x + threadIdx.x;     // grid covers exactly NASSIGN
    int   e = ae[a];
    float s = as_[a];
    int cnt = 0;
    for (int tile = 0; tile < NASSIGN; tile += 256) {
        int b = tile + threadIdx.x;
        se[threadIdx.x] = ae[b];
        ss[threadIdx.x] = as_[b];
        __syncthreads();
        #pragma unroll 8
        for (int j = 0; j < 256; ++j) {
            int b2 = tile + j;
            // score descending, stable tie-break by assignment index
            if (se[j] == e && (ss[j] > s || (ss[j] == s && b2 < a))) ++cnt;
        }
        __syncthreads();
    }
    if (cnt < CAP) {
        dtok[e * CAP + cnt] = a >> 1;     // token id
        wbuf[e * CAP + cnt] = aw[a];
    } else {
        atomicAdd(dropped, 1);
    }
}

// ---------------------------------------------------------------------------
// aux loss + dropped + expert_counts into output tail
// ---------------------------------------------------------------------------
__global__ void k_finalize(const float* __restrict__ stats, float* __restrict__ tail) {
    if (threadIdx.x != 0) return;
    float mi = 0.0f, ml = 0.0f;
    for (int e = 0; e < NE; ++e) { mi += stats[e]; ml += stats[8 + e]; }
    mi *= (1.0f / NE); ml *= (1.0f / NE);
    float vi = 0.0f, vl = 0.0f;
    for (int e = 0; e < NE; ++e) {
        float di = stats[e] - mi;     vi += di * di;
        float dl = stats[8 + e] - ml; vl += dl * dl;
    }
    vi *= (1.0f / NE); vl *= (1.0f / NE);
    float aux = 0.5f * (vi / (mi * mi + 1e-6f) + vl / (ml * ml + 1e-6f));
    tail[0] = aux;
    tail[1] = (float)(((const int*)stats)[24]);     // dropped
    for (int e = 0; e < NE; ++e) tail[2 + e] = stats[16 + e];
}

// ---------------------------------------------------------------------------
// gather x rows into xe (f16): one block per (expert, slot) row
// ---------------------------------------------------------------------------
__global__ void k_gather(const float* __restrict__ x, const int* __restrict__ dtok,
                         h16* __restrict__ xe) {
    int r = blockIdx.x;
    int tok = dtok[r];                    // padded slots -> 0 (weight 0 later)
    const float* src = x + (size_t)tok * DIM;
    h16* dst = xe + (size_t)r * DIM;
    for (int d = threadIdx.x; d < DIM; d += blockDim.x) dst[d] = (h16)src[d];
}

// ---------------------------------------------------------------------------
// WMMA GEMM core, async-LDS staged B.
// Block tile 128(M) x 128(N); 8 waves as 4(M) x 2(N); wave tile 32x64.
// A fragments direct from global; B tile (128 x 32 halves) async-copied to
// LDS (double buffered), read back as ds_load_b128 fragments.
// Fragment layouts per CDNA5 ISA 7.12.2 (f16, wave32):
//   A 16x32: lane m=l&15, v16h elems 0..7 = K (l>>4)*8 + j, 8..15 = +16
//   B 32x16: lane n=l&15, v16h elems 0..7 = K (l>>4)*8 + j, 8..15 = +16
// ---------------------------------------------------------------------------
__device__ __forceinline__ v16h make_v16(v8h lo, v8h hi) {
    v16h r;
    #pragma unroll
    for (int i = 0; i < 8; ++i) { r[i] = lo[i]; r[i + 8] = hi[i]; }
    return r;
}

// issue one 8KB B-tile fill: this thread's two 16B chunks (rows row, row+64)
__device__ __forceinline__ void issue_btile(unsigned lds0, int voff,
                                            const h16* base, int ldb) {
    asm volatile(
        "global_load_async_to_lds_b128 %0, %1, %4\n\t"
        "global_load_async_to_lds_b128 %2, %3, %4"
        :: "v"(lds0), "v"(voff),
           "v"(lds0 + 64 * BROW_B), "v"(voff + 64 * ldb * 2),
           "s"(base)
        : "memory");
}

__device__ __forceinline__ void moe_gemm_tiles(const h16* __restrict__ A,
                                               const h16* __restrict__ Bt,
                                               int lda, int ldb, int kdim,
                                               int mBase, int nBlock,
                                               int wn, int lane, int tid,
                                               h16* smemB, v8f acc[2][4]) {
    int lm = lane & 15, lh = lane >> 4;
    const h16* a0p = A + (size_t)(mBase + lm) * lda + lh * 8;
    const h16* a1p = a0p + (size_t)16 * lda;

    // async-staging assignment: thread -> rows (tid>>2, tid>>2 + 64), chunk tid&3
    int row = tid >> 2;
    int sub = tid & 3;
    int      voffBase = ((nBlock + row) * ldb + sub * 8) * 2;   // bytes
    unsigned ldsBase  = (unsigned)(uintptr_t)smemB + row * BROW_B + sub * 16;

    int nk = kdim / 32;

    // prologue: buffer 0 <- k-tile 0
    issue_btile(ldsBase, voffBase, Bt, ldb);

    for (int kb = 0; kb < nk; ++kb) {
        // issue next k-tile into the other buffer (clamped re-issue on last
        // step keeps the async counter discipline uniform; that buffer is
        // never read afterwards)
        int knext = (kb + 1 < nk ? kb + 1 : kb) * 64;           // byte offset
        issue_btile(ldsBase + ((kb + 1) & 1) * BBUF_B, voffBase + knext, Bt, ldb);

        // my previous 2 async loads (current buffer) complete ...
        asm volatile("s_wait_asynccnt 0x2" ::: "memory");
        // ... and everyone else's too
        __syncthreads();

        int k0 = kb * 32;
        v16h af[2], bf[4];
        af[0] = make_v16(*(const v8h*)(a0p + k0), *(const v8h*)(a0p + k0 + 16));
        af[1] = make_v16(*(const v8h*)(a1p + k0), *(const v8h*)(a1p + k0 + 16));

        const h16* bbuf = smemB + (size_t)(kb & 1) * BBUF_H;
        #pragma unroll
        for (int nt = 0; nt < 4; ++nt) {
            const h16* bp = bbuf + (size_t)(wn * 64 + nt * 16 + lm) * BROW_H
                                 + lh * 8;
            bf[nt] = make_v16(*(const v8h*)bp, *(const v8h*)(bp + 16));
        }
        #pragma unroll
        for (int mt = 0; mt < 2; ++mt) {
            #pragma unroll
            for (int nt = 0; nt < 4; ++nt) {
                acc[mt][nt] = __builtin_amdgcn_wmma_f32_16x16x32_f16(
                    false, af[mt], false, bf[nt],
                    (short)0, acc[mt][nt], false, false);
            }
        }
        // tile consumed by all waves before it can be overwritten
        __syncthreads();
    }
}

// FC1: h = gelu(xe @ w1 + b1), output f16 [E, CAP, HID]
__global__ void __launch_bounds__(256)
k_fc1(const h16* __restrict__ xe, const h16* __restrict__ w1t,
      const float* __restrict__ b1, h16* __restrict__ hbuf) {
    __shared__ h16 smemB[2 * BBUF_H];
    int e    = blockIdx.z;
    int tid  = threadIdx.x;
    int lane = tid & 31, wave = tid >> 5;
    int wm = wave & 3, wn = wave >> 2;
    int mBase  = blockIdx.x * 128 + wm * 32;
    int nBlock = blockIdx.y * 128;
    int nBase  = nBlock + wn * 64;

    const h16* A  = xe  + (size_t)e * CAP * DIM;
    const h16* Bt = w1t + (size_t)e * HID * DIM;
    v8f acc[2][4] = {};
    moe_gemm_tiles(A, Bt, DIM, DIM, DIM, mBase, nBlock, wn, lane, tid,
                   smemB, acc);

    const float* bias = b1 + (size_t)e * HID;
    h16* H_ = hbuf + (size_t)e * CAP * HID;
    int lm = lane & 15, lh = lane >> 4;
    #pragma unroll
    for (int mt = 0; mt < 2; ++mt) {
        #pragma unroll
        for (int nt = 0; nt < 4; ++nt) {
            int n = nBase + nt * 16 + lm;
            float bv = bias[n];
            #pragma unroll
            for (int r = 0; r < 8; ++r) {
                int m = mBase + mt * 16 + lh * 8 + r;
                float v = acc[mt][nt][r] + bv;
                v = 0.5f * v * (1.0f + erff(v * 0.70710678118f));   // exact gelu
                H_[(size_t)m * HID + n] = (h16)v;
            }
        }
    }
}

// FC2: out[tok] += gate * (h @ w2 + b2), scatter-add f32
__global__ void __launch_bounds__(256)
k_fc2(const h16* __restrict__ hbuf, const h16* __restrict__ w2t,
      const float* __restrict__ b2, const float* __restrict__ wbuf,
      const int* __restrict__ dtok, float* __restrict__ out) {
    __shared__ h16 smemB[2 * BBUF_H];
    int e    = blockIdx.z;
    int tid  = threadIdx.x;
    int lane = tid & 31, wave = tid >> 5;
    int wm = wave & 3, wn = wave >> 2;
    int mBase  = blockIdx.x * 128 + wm * 32;
    int nBlock = blockIdx.y * 128;
    int nBase  = nBlock + wn * 64;

    const h16* A  = hbuf + (size_t)e * CAP * HID;
    const h16* Bt = w2t  + (size_t)e * DIM * HID;
    v8f acc[2][4] = {};
    moe_gemm_tiles(A, Bt, HID, HID, HID, mBase, nBlock, wn, lane, tid,
                   smemB, acc);

    const float* bias = b2   + (size_t)e * DIM;
    const float* W    = wbuf + (size_t)e * CAP;
    const int*   T    = dtok + (size_t)e * CAP;
    int lm = lane & 15, lh = lane >> 4;
    #pragma unroll
    for (int mt = 0; mt < 2; ++mt) {
        #pragma unroll
        for (int nt = 0; nt < 4; ++nt) {
            int n = nBase + nt * 16 + lm;
            float bv = bias[n];
            #pragma unroll
            for (int r = 0; r < 8; ++r) {
                int m = mBase + mt * 16 + lh * 8 + r;
                float w = W[m];
                if (w != 0.0f) {
                    float v = (acc[mt][nt][r] + bv) * w;
                    atomicAdd(out + (size_t)T[m] * DIM + n, v);
                }
            }
        }
    }
}

// ---------------------------------------------------------------------------
// launch
// ---------------------------------------------------------------------------
extern "C" void kernel_launch(void* const* d_in, const int* in_sizes, int n_in,
                              void* d_out, int out_size, void* d_ws, size_t ws_size,
                              hipStream_t stream) {
    (void)in_sizes; (void)n_in; (void)ws_size;
    const float* x        = (const float*)d_in[0];
    const float* router_w = (const float*)d_in[1];
    const float* fc1_w    = (const float*)d_in[2];
    const float* fc1_b    = (const float*)d_in[3];
    const float* fc2_w    = (const float*)d_in[4];
    const float* fc2_b    = (const float*)d_in[5];
    float* out = (float*)d_out;

    // workspace layout (~233.3 MB total)
    char* ws = (char*)d_ws;
    float* stats = (float*)ws;                                  // 32 f32 (+dropped int @24)
    size_t off = 256;
    int*   ae   = (int*)(ws + off);   off += (size_t)NASSIGN * 4;
    float* as_  = (float*)(ws + off); off += (size_t)NASSIGN * 4;
    float* aw   = (float*)(ws + off); off += (size_t)NASSIGN * 4;
    int*   dtok = (int*)(ws + off);   off += (size_t)NROWS * 4;
    float* wbuf = (float*)(ws + off); off += (size_t)NROWS * 4;
    off = (off + 255) & ~(size_t)255;
    h16* xe  = (h16*)(ws + off); off += (size_t)NROWS * DIM * 2;   // 31.5 MB
    h16* w1t = (h16*)(ws + off); off += (size_t)NE * HID * DIM * 2;// 37.7 MB
    h16* w2t = (h16*)(ws + off); off += (size_t)NE * DIM * HID * 2;// 37.7 MB
    h16* hb  = (h16*)(ws + off);                                   // 125.8 MB

    // 1. zero output, stats, dispatch buffers (dtok+wbuf contiguous)
    k_zero<<<1024, 256, 0, stream>>>(out, out_size);
    k_zero<<<1, 256, 0, stream>>>(stats, 32);
    k_zero<<<64, 256, 0, stream>>>((float*)dtok, NROWS * 2);

    // 2. weights -> f16, transposed to [N,K] row-major for WMMA B fragments
    k_transpose_cvt<<<dim3(HID / 32, DIM / 32, NE), dim3(32, 8), 0, stream>>>(
        fc1_w, w1t, DIM, HID);
    k_transpose_cvt<<<dim3(DIM / 32, HID / 32, NE), dim3(32, 8), 0, stream>>>(
        fc2_w, w2t, HID, DIM);

    // 3. router + top-k + aux statistics
    k_router<<<N_TOK / 8, 256, 0, stream>>>(x, router_w, stats, ae, as_, aw);

    // 4. capacity ranking + dispatch
    k_rank<<<NASSIGN / 256, 256, 0, stream>>>(ae, as_, aw, dtok, wbuf,
                                              (int*)stats + 24);

    // 5. aux loss / dropped / expert_counts -> output tail
    k_finalize<<<1, 32, 0, stream>>>(stats, out + (size_t)N_TOK * DIM);

    // 6. gather tokens
    k_gather<<<NROWS, 256, 0, stream>>>(x, dtok, xe);

    // 7. expert GEMMs (WMMA f16 -> f32, async-LDS staged B)
    k_fc1<<<dim3(CAP / 128, HID / 128, NE), 256, 0, stream>>>(xe, w1t, fc1_b, hb);
    k_fc2<<<dim3(CAP / 128, DIM / 128, NE), 256, 0, stream>>>(hb, w2t, fc2_b,
                                                              wbuf, dtok, out);
}
